// M_GFAM_35948876268460
// MI455X (gfx1250) — matrix operand order repaired and checked by reading, not compiled, not run on hardware
//
#include <hip/hip_runtime.h>
#include <cstdint>
#include <cstddef>

// ---------------------------------------------------------------------------
// Problem constants (match reference)
// ---------------------------------------------------------------------------
#define NND   65536      // N = 256*256 nodes
#define CCHN  64         // channels
#define NEDG  524288     // E = 8*N
#define IMGW  256
#define NHEAD 4
#define DHEAD 16

typedef __attribute__((ext_vector_type(16))) _Float16 v16h;
typedef __attribute__((ext_vector_type(8)))  _Float16 v8h;
typedef __attribute__((ext_vector_type(8)))  float    v8f;

// D = A(16x32 f16) * B(32x16 f16) + C(16x16 f32)  -> v_wmma_f32_16x16x32_f16
__device__ __forceinline__ v8f wmma32(v16h a, v16h b, v8f c) {
  return __builtin_amdgcn_wmma_f32_16x16x32_f16(
      /*neg_a=*/false, a, /*neg_b=*/false, b,
      /*c_mod=*/(short)0, c, /*reuse_a=*/false, /*reuse_b=*/false);
}

__device__ __forceinline__ v16h cat8(v8h lo, v8h hi) {
  return __builtin_shufflevector(lo, hi, 0, 1, 2, 3, 4, 5, 6, 7,
                                 8, 9, 10, 11, 12, 13, 14, 15);
}

// ---------------------------------------------------------------------------
// Utility kernels
// ---------------------------------------------------------------------------
__global__ void k_fill(uint32_t* __restrict__ p, uint32_t v, int n) {
  int i = blockIdx.x * blockDim.x + threadIdx.x;
  if (i < n) p[i] = v;
}

// x [C,H,W] -> node-major [N,C]
__global__ void k_chw2nc(const float* __restrict__ X, float* __restrict__ Y) {
  int t = blockIdx.x * blockDim.x + threadIdx.x;       // t in [0, 64*N)
  int c = t >> 16;
  int p = t & (NND - 1);
  Y[(size_t)p * CCHN + c] = X[t];
}

// node-major [N,C] -> out [C,H,W]
__global__ void k_nc2chw(const float* __restrict__ X, float* __restrict__ Y) {
  int t = blockIdx.x * blockDim.x + threadIdx.x;
  int c = t >> 16;
  int p = t & (NND - 1);
  Y[t] = X[(size_t)p * CCHN + c];
}

// ---------------------------------------------------------------------------
// Weight pre-packing into WMMA B fragments (f16, lane-major, 32B/lane).
// Fragment layout: P[(fid*32 + lane)*16 + e], B-layout K(e) = e + 16*(lane>>4),
// column n = tn*16 + (lane&15).
// ---------------------------------------------------------------------------
template <int K, bool TRANSB>
__global__ void k_pack_b(const float* __restrict__ B, _Float16* __restrict__ P) {
  int t = blockIdx.x * blockDim.x + threadIdx.x;   // 4*(K/32)*512
  if (t >= 4 * (K / 32) * 512) return;
  int e    = t & 15;
  int lane = (t >> 4) & 31;
  int fid  = t >> 9;                               // tn*(K/32) + kc
  int kc = fid % (K / 32);
  int tn = fid / (K / 32);
  int r = lane & 15, kh = lane >> 4;
  int k = kc * 32 + e + 16 * kh;
  int n = tn * 16 + r;
  float v = TRANSB ? B[(size_t)n * K + k] : B[(size_t)k * 64 + n];
  P[t] = (_Float16)v;
}

// conv weights OIHW [64][64][3][3] -> fragments indexed fid=(tn*9+tap)*2+kc
__global__ void k_pack_conv(const float* __restrict__ W, _Float16* __restrict__ P) {
  int t = blockIdx.x * blockDim.x + threadIdx.x;   // 72*512 = 36864
  if (t >= 36864) return;
  int e    = t & 15;
  int lane = (t >> 4) & 31;
  int fid  = t >> 9;
  int kc  = fid & 1;
  int tap = (fid >> 1) % 9;
  int tn  = (fid >> 1) / 9;
  int r = lane & 15, kh = lane >> 4;
  int k = kc * 32 + e + 16 * kh;                   // input channel
  P[t] = (_Float16)W[(((size_t)(tn * 16 + r)) * CCHN + k) * 9 + tap];
}

// ---------------------------------------------------------------------------
// WMMA GEMM: Y[N, 64] = A[N, K] * B  (+ bias + residual when FUSED)
// Block = 128 threads = 4 waves: one 16-row tile, waves cover the 4 col tiles.
// A tile staged once in LDS as f16 (padded rows); B read pre-packed.
// ---------------------------------------------------------------------------
template <int K, bool FUSED>
__global__ __launch_bounds__(128) void k_gemm(
    const float* __restrict__ A, int lda,
    const _Float16* __restrict__ BP,
    const float* __restrict__ bias,
    const float* __restrict__ resid,
    float* __restrict__ Y, int ldy) {
  constexpr int NK   = K / 32;
  constexpr int LDSK = K + 8;                      // row pad: keep 16B align, skew banks
  __shared__ _Float16 sA[16 * LDSK];

  int tid  = threadIdx.x;
  int wid  = tid >> 5;
  int lane = tid & 31;
  int tm = blockIdx.x, tn = wid;
  int row0 = tm * 16;

  // stage + convert A tile (16 x K) once
  for (int t = tid; t < 16 * K; t += 128) {
    int row = t / K, k = t % K;
    sA[row * LDSK + k] = (_Float16)A[(size_t)(row0 + row) * lda + k];
  }
  __syncthreads();

  int r = lane & 15, kh = lane >> 4;
  v8f c = {};
#pragma unroll
  for (int kc = 0; kc < NK; ++kc) {
    const _Float16* ap = sA + r * LDSK + kc * 32 + 16 * kh;
    v16h a = cat8(*(const v8h*)ap, *(const v8h*)(ap + 8));
    v16h b = *(const v16h*)(BP + ((size_t)(tn * NK + kc) * 32 + lane) * 16);
    c = wmma32(a, b, c);
  }

  int col = tn * 16 + r;
  float bv = FUSED ? bias[col] : 0.0f;
#pragma unroll
  for (int e = 0; e < 8; ++e) {
    int row = row0 + e + 8 * kh;
    float v = c[e] + bv;
    if (FUSED) v += resid[(size_t)row * CCHN + col];
    Y[(size_t)row * ldy + col] = v;
  }
}

// ---------------------------------------------------------------------------
// 3x3 SAME conv, 64->64 ch, 9 shifted WMMA taps (18 WMMAs per wave).
// Block = 128 threads: one 16-pixel tile (row-contiguous), 4 output-ch tiles.
// 3x18 halo pixel block staged in LDS as f16.  MODE 0: PReLU, 1: +residual.
// ---------------------------------------------------------------------------
template <int MODE>
__global__ __launch_bounds__(128) void k_conv3(
    const float* __restrict__ X,
    const _Float16* __restrict__ WP,
    const float* __restrict__ bias,
    const float* __restrict__ prelu_a,
    const float* __restrict__ resid,
    float* __restrict__ Y) {
  constexpr int LDSC = 72;                         // padded channel stride
  __shared__ _Float16 sA[3 * 18 * LDSC];

  int tid  = threadIdx.x;
  int wid  = tid >> 5;
  int lane = tid & 31;
  int tm = blockIdx.x, tn = wid;
  int ytile = tm >> 4;                             // image row
  int xbase = (tm & 15) << 4;                      // first pixel x

  // stage 3 rows x 18 pixels x 64 channels with zero halo
  for (int t = tid; t < 3 * 18 * CCHN; t += 128) {
    int k  = t & 63;
    int p  = t >> 6;                               // 0..53
    int lx = p % 18, ly = p / 18;
    int yy = ytile + ly - 1;
    int xx = xbase + lx - 1;
    bool ok = (yy >= 0) && (yy < IMGW) && (xx >= 0) && (xx < IMGW);
    float v = ok ? X[((size_t)(yy * IMGW + xx)) * CCHN + k] : 0.0f;
    sA[(ly * 18 + lx) * LDSC + k] = (_Float16)v;
  }
  __syncthreads();

  int r = lane & 15, kh = lane >> 4;
  v8f c = {};
#pragma unroll
  for (int tap = 0; tap < 9; ++tap) {
    const int dy = tap / 3;                        // staged row (dy-1 applied above)
    const int dx = tap % 3;
    const _Float16* base = sA + ((size_t)(dy * 18 + r + dx)) * LDSC;
#pragma unroll
    for (int kc = 0; kc < 2; ++kc) {
      const _Float16* ap = base + kc * 32 + 16 * kh;
      v16h a = cat8(*(const v8h*)ap, *(const v8h*)(ap + 8));
      v16h b = *(const v16h*)(WP + ((size_t)((tn * 9 + tap) * 2 + kc) * 32 + lane) * 16);
      c = wmma32(a, b, c);
    }
  }

  int col = tn * 16 + r;
  float bv = bias[col];
  float aa = (MODE == 0) ? prelu_a[0] : 0.0f;
#pragma unroll
  for (int e = 0; e < 8; ++e) {
    int px = ytile * IMGW + xbase + e + 8 * kh;
    float v = c[e] + bv;
    if (MODE == 0) {
      v = (v >= 0.0f) ? v : aa * v;                // PReLU
    } else {
      v += resid[(size_t)px * CCHN + col];         // residual add
    }
    Y[(size_t)px * CCHN + col] = v;
  }
}

// ---------------------------------------------------------------------------
// GATv2 edge kernels
// ---------------------------------------------------------------------------
__device__ __forceinline__ void atomicMaxFloat(float* addr, float val) {
  if (val >= 0.0f) atomicMax((int*)addr, __float_as_int(val));
  else             atomicMin((unsigned int*)addr, __float_as_uint(val));
}

// pass 1: logits[e,h] = sum_d leakyrelu(xr[dst]+xj[src], 0.2) * att; segment max
__global__ void k_gat_logits(const float* __restrict__ xl,
                             const float* __restrict__ xr,
                             const int* __restrict__ ei,
                             const float* __restrict__ att,
                             float* __restrict__ logits,
                             float* __restrict__ mbuf) {
  int t = blockIdx.x * blockDim.x + threadIdx.x;
  if (t >= NEDG * NHEAD) return;
  int e = t >> 2, h = t & 3;
  int s = ei[e];
  int d = ei[NEDG + e];
  const float* pl = xl + (size_t)s * CCHN + h * DHEAD;
  const float* pr = xr + (size_t)d * CCHN + h * DHEAD;
  const float* pa = att + h * DHEAD;
  float acc = 0.0f;
#pragma unroll
  for (int k = 0; k < DHEAD; ++k) {
    float v = pr[k] + pl[k];
    v = (v >= 0.0f) ? v : 0.2f * v;
    acc += v * pa[k];
  }
  logits[t] = acc;
  atomicMaxFloat(&mbuf[d * NHEAD + h], acc);
}

// pass 2: ex = exp(logit - m[dst]) ; den[dst] += ex  (m == -inf -> 0)
__global__ void k_gat_expsum(const int* __restrict__ ei,
                             float* __restrict__ logits,
                             const float* __restrict__ mbuf,
                             float* __restrict__ den) {
  int t = blockIdx.x * blockDim.x + threadIdx.x;
  if (t >= NEDG * NHEAD) return;
  int e = t >> 2, h = t & 3;
  int d = ei[NEDG + e];
  float mv = mbuf[d * NHEAD + h];
  if (isinf(mv)) mv = 0.0f;
  float ex = expf(logits[t] - mv);
  logits[t] = ex;
  atomicAdd(&den[d * NHEAD + h], ex);
}

// pass 3: agg[dst] += xl[src] * ex/(den+1e-16)
__global__ void k_gat_agg(const float* __restrict__ xl,
                          const int* __restrict__ ei,
                          const float* __restrict__ ex,
                          const float* __restrict__ den,
                          float* __restrict__ agg) {
  int t = blockIdx.x * blockDim.x + threadIdx.x;
  if (t >= NEDG * NHEAD) return;
  int e = t >> 2, h = t & 3;
  int s = ei[e];
  int d = ei[NEDG + e];
  float alpha = ex[t] / (den[d * NHEAD + h] + 1e-16f);
  const float* ps = xl + (size_t)s * CCHN + h * DHEAD;
  float* pd = agg + (size_t)d * CCHN + h * DHEAD;
#pragma unroll
  for (int k = 0; k < DHEAD; ++k) atomicAdd(&pd[k], ps[k] * alpha);
}

// y = elu(agg + b), written to dest with row stride ldd and column offset coff
__global__ void k_bias_elu(const float* __restrict__ agg,
                           const float* __restrict__ b,
                           float* __restrict__ dest, int ldd, int coff) {
  int t = blockIdx.x * blockDim.x + threadIdx.x;   // [0, N*64)
  int i = t >> 6, c = t & 63;
  float v = agg[t] + b[c];
  v = (v > 0.0f) ? v : (expf(v) - 1.0f);
  dest[(size_t)i * ldd + coff + c] = v;
}

// ---------------------------------------------------------------------------
// Host-side orchestration
// ---------------------------------------------------------------------------
extern "C" void kernel_launch(void* const* d_in, const int* in_sizes, int n_in,
                              void* d_out, int out_size, void* d_ws, size_t ws_size,
                              hipStream_t stream) {
  (void)in_sizes; (void)n_in; (void)out_size; (void)ws_size;

  const float* X    = (const float*)d_in[0];   // [1,64,256,256]
  const int*   ei   = (const int*)  d_in[1];   // [2,E]
  const float* gWl  = (const float*)d_in[2];   // [9,3,64,64]
  const float* gWr  = (const float*)d_in[3];   // [9,3,64,64]
  const float* gatt = (const float*)d_in[4];   // [9,3,4,16]
  const float* gb   = (const float*)d_in[5];   // [9,3,64]
  const float* ccw  = (const float*)d_in[6];   // [3,64,192]
  const float* ccb  = (const float*)d_in[7];   // [3,64]
  const float* rbw  = (const float*)d_in[8];   // [2,4,2,64,64,3,3]
  const float* rbb  = (const float*)d_in[9];   // [2,4,2,64]
  const float* rba  = (const float*)d_in[10];  // [2,4]

  // workspace layout (floats, all region sizes are multiples of 32B)
  float* img    = (float*)d_ws;                     // [N,64]  current image
  float* cat    = img    + (size_t)NND * CCHN;      // [N,192] concat buffer
  float* xbuf   = cat    + (size_t)NND * 192;       // [N,64]  GAM running x
  float* xl     = xbuf   + (size_t)NND * CCHN;      // [N,64]
  float* xr     = xl     + (size_t)NND * CCHN;      // [N,64]
  float* agg    = xr     + (size_t)NND * CCHN;      // [N,64]
  float* hbuf   = agg    + (size_t)NND * CCHN;      // [N,64]  conv temp
  float* logits = hbuf   + (size_t)NND * CCHN;      // [E,4]
  float* mbuf   = logits + (size_t)NEDG * NHEAD;    // [N,4]
  float* den    = mbuf   + (size_t)NND * NHEAD;     // [N,4]
  _Float16* pkGat  = (_Float16*)(den + (size_t)NND * NHEAD); // 54 * 4096 halfs
  _Float16* pkCc   = pkGat  + (size_t)54 * 4096;             // 3 * 12288
  _Float16* pkConv = pkCc   + (size_t)3 * 12288;             // 16 * 36864

  const int GT = 256;
  const int TILE_BLK = 4096;                        // one 16-row tile per block
  const int ELT_BLK  = (NND * CCHN) / GT;           // 16384
  const int EDG_BLK  = (NEDG * NHEAD + GT - 1) / GT;// 8192
  const int NH_BLK   = (NND * NHEAD) / GT;          // 1024

  // ---- pre-pack all weights into f16 WMMA fragments (L2-resident) ----
  for (int ml = 0; ml < 27; ++ml) {
    k_pack_b<64, false><<<16, GT, 0, stream>>>(gWl + (size_t)ml * 4096,
                                               pkGat + (size_t)(2 * ml) * 4096);
    k_pack_b<64, false><<<16, GT, 0, stream>>>(gWr + (size_t)ml * 4096,
                                               pkGat + (size_t)(2 * ml + 1) * 4096);
  }
  for (int s = 0; s < 3; ++s)
    k_pack_b<192, true><<<48, GT, 0, stream>>>(ccw + (size_t)s * CCHN * 192,
                                               pkCc + (size_t)s * 12288);
  for (int j = 0; j < 16; ++j)
    k_pack_conv<<<144, GT, 0, stream>>>(rbw + (size_t)j * CCHN * CCHN * 9,
                                        pkConv + (size_t)j * 36864);

  // image -> node-major
  k_chw2nc<<<ELT_BLK, GT, 0, stream>>>(X, img);

  for (int s = 0; s < 3; ++s) {
    // --- 3 GraphAttentionModule branches -> cat[N,192] ---
    for (int t = 0; t < 3; ++t) {
      int m = s * 3 + t;
      const float* xin = img;
      for (int l = 0; l < 3; ++l) {
        int ml = m * 3 + l;
        k_gemm<64, false><<<TILE_BLK, 128, 0, stream>>>(
            xin, CCHN, pkGat + (size_t)(2 * ml) * 4096, nullptr, nullptr, xl, CCHN);
        k_gemm<64, false><<<TILE_BLK, 128, 0, stream>>>(
            xin, CCHN, pkGat + (size_t)(2 * ml + 1) * 4096, nullptr, nullptr, xr, CCHN);
        k_fill<<<NH_BLK, GT, 0, stream>>>((uint32_t*)mbuf, 0xFF800000u, NND * NHEAD); // -inf
        k_fill<<<NH_BLK, GT, 0, stream>>>((uint32_t*)den, 0u, NND * NHEAD);
        k_fill<<<ELT_BLK, GT, 0, stream>>>((uint32_t*)agg, 0u, NND * CCHN);
        k_gat_logits<<<EDG_BLK, GT, 0, stream>>>(xl, xr, ei, gatt + (size_t)ml * 64,
                                                 logits, mbuf);
        k_gat_expsum<<<EDG_BLK, GT, 0, stream>>>(ei, logits, mbuf, den);
        k_gat_agg<<<EDG_BLK, GT, 0, stream>>>(xl, ei, logits, den, agg);
        if (l == 2) {
          k_bias_elu<<<ELT_BLK, GT, 0, stream>>>(agg, gb + (size_t)ml * CCHN,
                                                 cat, 192, t * CCHN);
        } else {
          k_bias_elu<<<ELT_BLK, GT, 0, stream>>>(agg, gb + (size_t)ml * CCHN,
                                                 xbuf, CCHN, 0);
          xin = xbuf;
        }
      }
    }
    // --- concat 1x1 conv: img = cat @ ccw[s]^T + ccb[s] + img ---
    k_gemm<192, true><<<TILE_BLK, 128, 0, stream>>>(
        cat, 192, pkCc + (size_t)s * 12288,
        ccb + (size_t)s * CCHN, img, img, CCHN);
    // --- resblock stack (after stages 0 and 1 only) ---
    if (s < 2) {
      for (int blk = 0; blk < 4; ++blk) {
        int j0 = (s * 4 + blk) * 2;
        size_t bbase = (size_t)j0 * CCHN;
        const float* a = rba + (s * 4 + blk);
        k_conv3<0><<<TILE_BLK, 128, 0, stream>>>(
            img, pkConv + (size_t)j0 * 36864, rbb + bbase, a, nullptr, hbuf);
        k_conv3<1><<<TILE_BLK, 128, 0, stream>>>(
            hbuf, pkConv + (size_t)(j0 + 1) * 36864, rbb + bbase + CCHN,
            nullptr, img, img);
      }
    }
  }

  // node-major -> output [1,64,256,256]
  k_nc2chw<<<ELT_BLK, GT, 0, stream>>>(img, (float*)d_out);
}
